// MostSimilar_11536282157562
// MI455X (gfx1250) — matrix operand order repaired and checked by reading, not compiled, not run on hardware
//
#include <hip/hip_runtime.h>
#include <cmath>

// ---------------------------------------------------------------------------
// train: [B=4, K=64, N=32, D=128] f32 -> per batch 2048 x 128
// test : [B=4, M=4096, D=128]     f32
// out  : labels [B=4, M=4096] = argmax_k max_n cos(te_m, tr_kn)
//
// Phase 1: L2-normalize to f16 in workspace (K-contiguous).
// Phase 2: V_WMMA_F32_16X16X32_F16 GEMM tiles, register double-buffered B
//          fragments, per-lane deferred argmax (no ds ops in the hot loop).
// ---------------------------------------------------------------------------

typedef _Float16 v16h __attribute__((ext_vector_type(16)));
typedef _Float16 v8h  __attribute__((ext_vector_type(8)));
typedef _Float16 v4h  __attribute__((ext_vector_type(4)));
typedef float    v8f  __attribute__((ext_vector_type(8)));

#define D_DIM   128
#define B_BATCH 4
#define M_TEST  4096
#define N_TRAIN 2048   // 64 classes * 32 shots
#define K_CLS   64
#define N_SHOT  32

// Compare+select max: inputs are guaranteed finite (normalized), so this is
// equivalent to fmaxf but avoids the backend's v_max_num canonicalization
// (v_max x,y,y self-max ops) -> v_cmp + v_cndmask, dual-issueable.
__device__ __forceinline__ float smax(float a, float b) { return a > b ? a : b; }

// --------------------------- normalization kernel --------------------------
__global__ __launch_bounds__(256) void norm_rows_f16(const float* __restrict__ src,
                                                     _Float16* __restrict__ dst,
                                                     int nrows) {
  const int wid  = threadIdx.x >> 5;
  const int lane = threadIdx.x & 31;
  const int row  = blockIdx.x * (blockDim.x >> 5) + wid;
  if (row >= nrows) return;

  const float4 v = *(const float4*)(src + (size_t)row * D_DIM + lane * 4);
  float ss = v.x * v.x + v.y * v.y + v.z * v.z + v.w * v.w;
#pragma unroll
  for (int off = 16; off > 0; off >>= 1) ss += __shfl_xor(ss, off, 32);
  float nrm = sqrtf(ss);
  if (nrm < 1e-8f) nrm = 1e-8f;
  const float inv = 1.0f / nrm;

  v4h h;
  h[0] = (_Float16)(v.x * inv);
  h[1] = (_Float16)(v.y * inv);
  h[2] = (_Float16)(v.z * inv);
  h[3] = (_Float16)(v.w * inv);
  *(v4h*)(dst + (size_t)row * D_DIM + lane * 4) = h;
}

// ------------------------------ WMMA kernel --------------------------------
// Block = 128 threads = 4 waves; each wave owns one 16-row m-tile.
__global__ __launch_bounds__(128) void knn_wmma(const _Float16* __restrict__ te,
                                                const _Float16* __restrict__ tr,
                                                float* __restrict__ out) {
  const int wid  = threadIdx.x >> 5;
  const int lane = threadIdx.x & 31;
  const int half = lane >> 4;   // lane half selects 16-wide K sub-chunk (contiguous k = 16*half + s, A and B alike)
  const int lid  = lane & 15;   // A: row M within tile; B: column N within tile

  const int b      = blockIdx.x >> 6;   // 64 blocks per batch
  const int mblock = blockIdx.x & 63;
  const int m0     = mblock * 64 + wid * 16;

  // ---- A fragments: 16 test rows x K=128, resident in 32 VGPRs ----
  const _Float16* teRow = te + (size_t)(b * M_TEST + m0 + lid) * D_DIM;
  v16h afrag[4];
#pragma unroll
  for (int kc = 0; kc < 4; ++kc) {
    const int k0 = kc * 32;
    const v8h lo = *(const v8h*)(teRow + k0 + half * 16);
    const v8h hi = *(const v8h*)(teRow + k0 + half * 16 + 8);
    afrag[kc] = __builtin_shufflevector(lo, hi,
        0, 1, 2, 3, 4, 5, 6, 7, 8, 9, 10, 11, 12, 13, 14, 15);
  }

  const _Float16* trBatch = tr + (size_t)b * N_TRAIN * D_DIM;

  // B fragment loader: class k -> 8 v16h (two 16-column subtiles x 4 K-chunks).
  auto loadB = [&](int k, v16h (&dst)[8]) {
    const _Float16* r0 = trBatch + (size_t)(k * N_SHOT + lid) * D_DIM + half * 16;
    const _Float16* r1 = r0 + (size_t)16 * D_DIM;
#pragma unroll
    for (int kc = 0; kc < 4; ++kc) {
      dst[kc]     = *(const v16h*)(r0 + kc * 32);
      dst[4 + kc] = *(const v16h*)(r1 + kc * 32);
    }
  };

  // Per-lane running argmax (lane sees shots {lid, lid+16} of every class);
  // cross-lane combine deferred to the epilogue.
  float bestv[8];
  int   besti[8];
#pragma unroll
  for (int v = 0; v < 8; ++v) { bestv[v] = -1e30f; besti[v] = 0; }

  auto process = [&](int k, const v16h (&buf)[8]) {
    v8f c0 = {}, c1 = {};
#pragma unroll
    for (int kc = 0; kc < 4; ++kc)
      c0 = __builtin_amdgcn_wmma_f32_16x16x32_f16(false, afrag[kc], false, buf[kc],
                                                  (short)0, c0, false, false);
#pragma unroll
    for (int kc = 0; kc < 4; ++kc)
      c1 = __builtin_amdgcn_wmma_f32_16x16x32_f16(false, afrag[kc], false, buf[4 + kc],
                                                  (short)0, c1, false, false);
#pragma unroll
    for (int v = 0; v < 8; ++v) {
      const float cm = smax(c0[v], c1[v]);                 // shots lid, lid+16
      if (cm > bestv[v]) { bestv[v] = cm; besti[v] = k; }  // first-k tie-break
    }
  };

  // Software-pipelined class loop: next class's 16 b128 loads are in flight
  // while the current (resident) buffer feeds the matrix pipe.
  v16h bufA[8], bufB[8];
  loadB(0, bufA);
  for (int k = 0; k < K_CLS; k += 2) {
    loadB(k + 1, bufB);
    if (k + 2 < K_CLS)
      __builtin_prefetch(trBatch + (size_t)((k + 2) * N_SHOT + lid) * D_DIM, 0, 0);
    process(k, bufA);
    if (k + 2 < K_CLS) loadB(k + 2, bufA);
    process(k + 1, bufB);
  }

  // Epilogue: cross-lane argmax over the 16 columns of each half.
  // Larger value wins; equal values -> smaller class index (jnp.argmax).
#pragma unroll
  for (int off = 1; off < 16; off <<= 1) {
#pragma unroll
    for (int v = 0; v < 8; ++v) {
      const float ov = __shfl_xor(bestv[v], off, 32);
      const int   oi = __shfl_xor(besti[v], off, 32);
      if (ov > bestv[v] || (ov == bestv[v] && oi < besti[v])) {
        bestv[v] = ov; besti[v] = oi;
      }
    }
  }

  if (lid == 0) {
    float* o = out + (size_t)b * M_TEST + m0 + half * 8;
#pragma unroll
    for (int v = 0; v < 8; ++v) o[v] = (float)besti[v];
  }
}

// ------------------------------- launcher ----------------------------------
extern "C" void kernel_launch(void* const* d_in, const int* in_sizes, int n_in,
                              void* d_out, int out_size, void* d_ws, size_t ws_size,
                              hipStream_t stream) {
  const float* tr_f32 = (const float*)d_in[0];  // [4,64,32,128]
  const float* te_f32 = (const float*)d_in[1];  // [4,4096,128]
  float*       out    = (float*)d_out;          // [4,4096] labels

  _Float16* teH = (_Float16*)d_ws;                        // 4 MB
  _Float16* trH = teH + (size_t)B_BATCH * M_TEST * D_DIM; // +2 MB

  const int te_rows = B_BATCH * M_TEST;   // 16384
  const int tr_rows = B_BATCH * N_TRAIN;  // 8192

  norm_rows_f16<<<te_rows / 8, 256, 0, stream>>>(te_f32, teH, te_rows);
  norm_rows_f16<<<tr_rows / 8, 256, 0, stream>>>(tr_f32, trH, tr_rows);

  knn_wmma<<<B_BATCH * (M_TEST / 64), 128, 0, stream>>>(teH, trH, out);
}
